// MultiheadAttention_59296318488705
// MI455X (gfx1250) — matrix operand order repaired and checked by reading, not compiled, not run on hardware
//
#include <hip/hip_runtime.h>

typedef __bf16 bf16;
typedef __attribute__((ext_vector_type(16))) __bf16 v16bf;
typedef __attribute__((ext_vector_type(8)))  __bf16 v8bf;
typedef __attribute__((ext_vector_type(8)))  float  v8f;
typedef __attribute__((ext_vector_type(4)))  unsigned int u32x4;
typedef __attribute__((ext_vector_type(8)))  unsigned int u32x8;

#define BB   2
#define SS   2048
#define DD   1024
#define HH   16
#define HDIM 64
#define MM   (BB * SS)   // 4096 rows of x / ctx

// ---------------------------------------------------------------------------
// fp32 -> bf16 elementwise convert
// ---------------------------------------------------------------------------
__global__ void cvt_f32_bf16(const float* __restrict__ in, bf16* __restrict__ out, int n) {
    int i = blockIdx.x * blockDim.x + threadIdx.x;
    if (i < n) out[i] = (bf16)in[i];
}

// ---------------------------------------------------------------------------
// W [K][N] fp32  ->  Wt [N][K] bf16   (so WMMA B-fragments are contiguous)
// ---------------------------------------------------------------------------
__global__ void transpose_bf16(const float* __restrict__ W, bf16* __restrict__ Wt) {
    int n = blockIdx.x * blockDim.x + threadIdx.x;   // 0..1023 (coalesced read)
    int k = blockIdx.y;                              // 0..1023
    Wt[(size_t)n * DD + k] = (bf16)W[(size_t)k * DD + n];
}

// ---------------------------------------------------------------------------
// GEMM: C[4096][1024] = A[4096][1024] * Bt[1024][1024]^T  (bf16 in, fp32 acc)
// Block = 8 waves, block tile 256(M) x 64(N); wave tile 32(M) x 64(N).
// B tile (64 rows x 32 k = 4KB) staged to LDS by the Tensor Data Mover,
// double buffered; wave 0 issues the TDM descriptor, everyone consumes after
// s_wait_tensorcnt + barrier.
// mode 0: out bf16 head-major  [b,h,s,64]   (Q, K)
// mode 1: out bf16 transposed  [b,h,64,s]   (V)
// mode 2: out fp32 row-major + bias         (final projection)
// ---------------------------------------------------------------------------
__global__ void __launch_bounds__(256)
gemm_bf16(const bf16* __restrict__ A, const bf16* __restrict__ Bt,
          int mode, bf16* __restrict__ out_b, float* __restrict__ out_f,
          const float* __restrict__ bias)
{
    __shared__ bf16 ldsB[2][64][32];     // 2 x 4KB double buffer

    const int wave = threadIdx.x >> 5;
    const int lane = threadIdx.x & 31;
    const int lh   = lane >> 4;          // half-wave
    const int lm   = lane & 15;

    const int mBase = blockIdx.y * 256 + wave * 32;
    const int nBase = blockIdx.x * 64;

    v8f acc[2][4] = {};

    const bf16* aRow0 = A + (size_t)(mBase + lm) * DD;
    const bf16* aRow1 = A + (size_t)(mBase + 16 + lm) * DD;

    // ---- TDM descriptor group 1 (constant): 2B elems, 64x32 tile, stride DD
    u32x8 g1;
    g1[0] = (1u << 16);                         // workgroup_mask=0, data_size=1 (2B)
    g1[1] = ((unsigned)DD & 0xFFFFu) << 16;     // tensor_dim0[15:0] @ bits 63:48
    g1[2] = (((unsigned)DD >> 16) & 0xFFFFu)    // tensor_dim0[31:16]
          | (((unsigned)DD & 0xFFFFu) << 16);   // tensor_dim1[15:0]
    g1[3] = (((unsigned)DD >> 16) & 0xFFFFu)    // tensor_dim1[31:16]
          | (32u << 16);                        // tile_dim0 = 32 elements
    g1[4] = 64u;                                // tile_dim1 = 64, tile_dim2 = 0
    g1[5] = (unsigned)DD;                       // tensor_dim0_stride[31:0]
    g1[6] = 0u;                                 // stride hi, dim1_stride lo
    g1[7] = 0u;

    const unsigned ldsOff0 = (unsigned)(uintptr_t)&ldsB[0][0][0];
    const unsigned ldsOff1 = (unsigned)(uintptr_t)&ldsB[1][0][0];

    auto tdm_issue = [&](int buf, int k0) {
        unsigned long long ga =
            (unsigned long long)(uintptr_t)(Bt + (size_t)nBase * DD + k0);
        u32x4 g0;
        g0[0] = 1u;                                        // count=1, user mode
        g0[1] = buf ? ldsOff1 : ldsOff0;                   // lds_addr
        g0[2] = (unsigned)(ga & 0xFFFFFFFFu);              // global_addr lo
        g0[3] = ((unsigned)(ga >> 32) & 0x01FFFFFFu)       // global_addr hi
              | 0x80000000u;                               // type = 2 ("image")
        asm volatile("tensor_load_to_lds %0, %1" :: "s"(g0), "s"(g1) : "memory");
    };

    if (wave == 0) tdm_issue(0, 0);
    int cur = 0;

    for (int k0 = 0; k0 < DD; k0 += 32) {
        const bool hasNext = (k0 + 32) < DD;
        if (wave == 0) {
            if (hasNext) {
                tdm_issue(cur ^ 1, k0 + 32);               // overlap next DMA
                __builtin_amdgcn_s_wait_tensorcnt(1);      // wait for `cur` only
            } else {
                __builtin_amdgcn_s_wait_tensorcnt(0);      // last tile: drain
            }
        }
        __syncthreads();                                   // buffer `cur` ready

        // A fragments (16x32): two contiguous 16B loads per lane per row-group
        union { v16bf v; v8bf h2[2]; } ua0, ua1;
        ua0.h2[0] = *(const v8bf*)(aRow0 + k0 + 16 * lh);
        ua0.h2[1] = *(const v8bf*)(aRow0 + k0 + 16 * lh + 8);
        ua1.h2[0] = *(const v8bf*)(aRow1 + k0 + 16 * lh);
        ua1.h2[1] = *(const v8bf*)(aRow1 + k0 + 16 * lh + 8);
        __builtin_prefetch(aRow0 + k0 + 128, 0, 1);        // global_prefetch_b8

        #pragma unroll
        for (int j = 0; j < 4; ++j) {
            // B fragment (32x16): lane = col n, 16 contiguous K per half-wave
            v16bf bv = *(const v16bf*)&ldsB[cur][j * 16 + lm][16 * lh];
            acc[0][j] = __builtin_amdgcn_wmma_f32_16x16x32_bf16(
                false, ua0.v, false, bv, (short)0, acc[0][j], false, false);
            acc[1][j] = __builtin_amdgcn_wmma_f32_16x16x32_bf16(
                false, ua1.v, false, bv, (short)0, acc[1][j], false, false);
        }
        __syncthreads();                                   // done reading `cur`
        cur ^= 1;
    }

    // C/D layout: VGPR r, lane -> element C[r + 8*lh][lm]
    #pragma unroll
    for (int g = 0; g < 2; ++g) {
        #pragma unroll
        for (int j = 0; j < 4; ++j) {
            #pragma unroll
            for (int r = 0; r < 8; ++r) {
                int m = mBase + g * 16 + r + 8 * lh;
                int n = nBase + j * 16 + lm;
                float v = acc[g][j][r];
                if (mode == 2) {
                    out_f[(size_t)m * DD + n] = v + bias[n];
                } else {
                    int b = m / SS, s = m % SS;
                    int h = n / HDIM, d = n % HDIM;
                    if (mode == 0)
                        out_b[(((size_t)(b * HH + h) * SS + s) * HDIM) + d] = (bf16)v;
                    else
                        out_b[(((size_t)(b * HH + h) * HDIM + d) * SS) + s] = (bf16)v;
                }
            }
        }
    }
}

// ---------------------------------------------------------------------------
// Causal flash attention.  One wave per (b, h, 16-query-row tile).
// Qh/Kh: [b,h,s,64] bf16.  Vt: [b,h,64,s] bf16.  ctx out: [b*s][1024] bf16.
// No block barriers (waves have different causal trip counts): the P staging
// LDS patch is wave-private, ordered with a bare s_wait_dscnt.
// ---------------------------------------------------------------------------
__global__ void __launch_bounds__(256)
attn_fwd(const bf16* __restrict__ Qh, const bf16* __restrict__ Kh,
         const bf16* __restrict__ Vt, bf16* __restrict__ ctxb)
{
    __shared__ bf16 lds_p[8][16 * 32];   // per-wave private P staging (1KB each)

    const int wave = threadIdx.x >> 5;
    const int lane = threadIdx.x & 31;
    const int lh   = lane >> 4;
    const int lm   = lane & 15;

    const int QT  = SS / 16;
    const int wid = blockIdx.x * 8 + wave;
    const int b   = wid / (HH * QT);
    const int rem = wid % (HH * QT);
    const int h   = rem / QT;
    const int q0  = (rem % QT) * 16;

    const bf16* Qb = Qh + (size_t)(b * HH + h) * SS * HDIM;
    const bf16* Kb = Kh + (size_t)(b * HH + h) * SS * HDIM;
    const bf16* Vb = Vt + (size_t)(b * HH + h) * HDIM * SS;

    // Q tile as two A-fragments (head-dim 0..31 and 32..63)
    const bf16* qRow = Qb + (size_t)(q0 + lm) * HDIM;
    union { v16bf v; v8bf h2[2]; } aq0, aq1;
    aq0.h2[0] = *(const v8bf*)(qRow + 0  + 16 * lh);
    aq0.h2[1] = *(const v8bf*)(qRow + 8  + 16 * lh);
    aq1.h2[0] = *(const v8bf*)(qRow + 32 + 16 * lh);
    aq1.h2[1] = *(const v8bf*)(qRow + 40 + 16 * lh);

    float mstate[8], lstate[8];
    #pragma unroll
    for (int r = 0; r < 8; ++r) { mstate[r] = -3.0e38f; lstate[r] = 0.0f; }
    v8f ctx[4] = {};

    const float scale = 0.125f;          // 1/sqrt(64)
    const int kmax = q0 + 16;            // causal limit (exclusive)

    for (int k0 = 0; k0 < kmax; k0 += 32) {
        const bool have1 = (k0 + 16) < kmax;

        // ---- S = Q * K^T for 32 keys (two 16x16 fragments) ----
        v8f s0v = {}, s1v = {};
        {
            const bf16* kRow = Kb + (size_t)(k0 + lm) * HDIM;     // key = k0 + lm
            v16bf bk00 = *(const v16bf*)(kRow + 16 * lh);         // d 0..31
            v16bf bk01 = *(const v16bf*)(kRow + 32 + 16 * lh);    // d 32..63
            s0v = __builtin_amdgcn_wmma_f32_16x16x32_bf16(false, aq0.v, false, bk00, (short)0, s0v, false, false);
            s0v = __builtin_amdgcn_wmma_f32_16x16x32_bf16(false, aq1.v, false, bk01, (short)0, s0v, false, false);
        }
        if (have1) {
            const bf16* kRow = Kb + (size_t)(k0 + 16 + lm) * HDIM;
            v16bf bk10 = *(const v16bf*)(kRow + 16 * lh);
            v16bf bk11 = *(const v16bf*)(kRow + 32 + 16 * lh);
            s1v = __builtin_amdgcn_wmma_f32_16x16x32_bf16(false, aq0.v, false, bk10, (short)0, s1v, false, false);
            s1v = __builtin_amdgcn_wmma_f32_16x16x32_bf16(false, aq1.v, false, bk11, (short)0, s1v, false, false);
        }

        // ---- hoist V fragment loads: softmax VALU below hides their latency
        v16bf bv[4];
        #pragma unroll
        for (int j = 0; j < 4; ++j)
            bv[j] = *(const v16bf*)(Vb + (size_t)(j * 16 + lm) * SS + k0 + 16 * lh);

        // ---- scale + causal mask + online softmax ----
        #pragma unroll
        for (int r = 0; r < 8; ++r) {
            const int q    = q0 + r + 8 * lh;
            const int key0 = k0 + lm;
            const int key1 = key0 + 16;
            float f0 = (key0 <= q) ? s0v[r] * scale : -3.0e38f;
            float f1 = (have1 && key1 <= q) ? s1v[r] * scale : -3.0e38f;

            float mt = fmaxf(f0, f1);
            mt = fmaxf(mt, __shfl_xor(mt, 1, 32));
            mt = fmaxf(mt, __shfl_xor(mt, 2, 32));
            mt = fmaxf(mt, __shfl_xor(mt, 4, 32));
            mt = fmaxf(mt, __shfl_xor(mt, 8, 32));

            const float mnew = fmaxf(mstate[r], mt);
            const float p0   = __expf(f0 - mnew);
            const float p1   = __expf(f1 - mnew);

            float rs = p0 + p1;
            rs += __shfl_xor(rs, 1, 32);
            rs += __shfl_xor(rs, 2, 32);
            rs += __shfl_xor(rs, 4, 32);
            rs += __shfl_xor(rs, 8, 32);

            const float corr = __expf(mstate[r] - mnew);
            lstate[r] = lstate[r] * corr + rs;
            mstate[r] = mnew;
            #pragma unroll
            for (int j = 0; j < 4; ++j) ctx[j][r] *= corr;

            const int row = r + 8 * lh;
            lds_p[wave][row * 32 + lm]      = (bf16)p0;
            lds_p[wave][row * 32 + 16 + lm] = (bf16)p1;
        }

        // wave-private LDS RAW across lanes: DS counter wait only, no barrier
        asm volatile("s_wait_dscnt 0" ::: "memory");

        union { v16bf v; v8bf h2[2]; } ap;
        const bf16* pr = &lds_p[wave][lm * 32];
        ap.h2[0] = *(const v8bf*)(pr + 16 * lh);
        ap.h2[1] = *(const v8bf*)(pr + 16 * lh + 8);

        // ---- ctx += P * V ----
        #pragma unroll
        for (int j = 0; j < 4; ++j)
            ctx[j] = __builtin_amdgcn_wmma_f32_16x16x32_bf16(
                false, ap.v, false, bv[j], (short)0, ctx[j], false, false);
    }

    // ---- normalize and store ctx (bf16, row-major [b*s][1024]) ----
    #pragma unroll
    for (int j = 0; j < 4; ++j) {
        #pragma unroll
        for (int r = 0; r < 8; ++r) {
            const int   q   = q0 + r + 8 * lh;
            const int   dim = j * 16 + lm;
            const float v   = ctx[j][r] / lstate[r];
            ctxb[((size_t)b * SS + q) * DD + h * HDIM + dim] = (bf16)v;
        }
    }
}

// ---------------------------------------------------------------------------
// Host-side launch
// ---------------------------------------------------------------------------
extern "C" void kernel_launch(void* const* d_in, const int* in_sizes, int n_in,
                              void* d_out, int out_size, void* d_ws, size_t ws_size,
                              hipStream_t stream) {
    (void)in_sizes; (void)n_in; (void)out_size; (void)ws_size;

    const float* x  = (const float*)d_in[0];
    const float* Wq = (const float*)d_in[1];
    const float* Wk = (const float*)d_in[2];
    const float* Wv = (const float*)d_in[3];
    const float* Wo = (const float*)d_in[4];
    const float* bo = (const float*)d_in[5];
    float* out = (float*)d_out;

    char*  ws  = (char*)d_ws;
    size_t off = 0;
    auto alloc = [&](size_t bytes) -> void* {
        void* p = ws + off;
        off += (bytes + 255) & ~(size_t)255;
        return p;
    };
    bf16* xb   = (bf16*)alloc((size_t)MM * DD * 2);
    bf16* wtq  = (bf16*)alloc((size_t)DD * DD * 2);
    bf16* wtk  = (bf16*)alloc((size_t)DD * DD * 2);
    bf16* wtv  = (bf16*)alloc((size_t)DD * DD * 2);
    bf16* wto  = (bf16*)alloc((size_t)DD * DD * 2);
    bf16* qh   = (bf16*)alloc((size_t)MM * DD * 2);
    bf16* kh   = (bf16*)alloc((size_t)MM * DD * 2);
    bf16* vt   = (bf16*)alloc((size_t)MM * DD * 2);
    bf16* ctxb = (bf16*)alloc((size_t)MM * DD * 2);

    const int ncvt = MM * DD;
    cvt_f32_bf16<<<ncvt / 256, 256, 0, stream>>>(x, xb, ncvt);

    dim3 tg(DD / 256, DD);
    transpose_bf16<<<tg, 256, 0, stream>>>(Wq, wtq);
    transpose_bf16<<<tg, 256, 0, stream>>>(Wk, wtk);
    transpose_bf16<<<tg, 256, 0, stream>>>(Wv, wtv);
    transpose_bf16<<<tg, 256, 0, stream>>>(Wo, wto);

    dim3 gg(DD / 64, MM / 256);
    gemm_bf16<<<gg, 256, 0, stream>>>(xb, wtq, 0, qh, nullptr, nullptr);
    gemm_bf16<<<gg, 256, 0, stream>>>(xb, wtk, 0, kh, nullptr, nullptr);
    gemm_bf16<<<gg, 256, 0, stream>>>(xb, wtv, 1, vt, nullptr, nullptr);

    const int nWaves = BB * HH * (SS / 16);
    attn_fwd<<<nWaves / 8, 256, 0, stream>>>(qh, kh, vt, ctxb);

    gemm_bf16<<<gg, 256, 0, stream>>>(ctxb, wto, 2, nullptr, out, bo);
}
